// NMNIST_LNN_19877108646019
// MI455X (gfx1250) — matrix operand order repaired and checked by reading, not compiled, not run on hardware
//
#include <hip/hip_runtime.h>

// CfC recurrent network, B=1024, T=2048, IN=4, UNITS=256, BB=128, NC=10.
// One workgroup (256 threads = 8 wave32) per 16-row batch tile -> 64 WGs.
// bf16 WMMA with f32 accumulation; head weights resident in VGPRs;
// W_ext (concat [x|h] weights) fragments resident in LDS; W_fc folded out
// of the time loop via mean(h) @ W_fc.
//
// Round-2 changes:
//  - Phase A issues all 36 ds_load_b128 up front (graduated dscnt waits
//    instead of a full drain before every wmma).
//  - x_{t+1} register-prefetched right after the barrier so the global load
//    latency is hidden behind a full step of compute.

#define B_SZ   1024
#define T_SZ   2048
#define IN_SZ  4
#define UNITS  256
#define BBD    128
#define NC     10
#define KEXT   288   // IN+UNITS = 260 padded to 9 K-tiles of 32

typedef __attribute__((ext_vector_type(16))) __bf16 v16bf;
typedef __attribute__((ext_vector_type(8)))  float  v8f;

union BF16Frag { v16bf v; __bf16 e[16]; uint4 q[2]; };

#if __has_builtin(__builtin_amdgcn_tanhf)
__device__ __forceinline__ float fast_tanh(float x) { return __builtin_amdgcn_tanhf(x); }
#else
__device__ __forceinline__ float fast_tanh(float x) { return tanhf(x); }
#endif

__device__ __forceinline__ v8f splat8(float s) {
    v8f v;
#pragma unroll
    for (int i = 0; i < 8; ++i) v[i] = s;
    return v;
}

// A-matrix fragment (16x32 bf16) from a row-major LDS buffer.
// Lane L (row = L&15, hi = L>>4): elems 0..15 = K kbase+hi*16..kbase+hi*16+15
__device__ __forceinline__ v16bf ldsA_frag(const __bf16* base, int rowStride, int kbase, int lane) {
    const int row = lane & 15, hi = lane >> 4;
    const __bf16* p = base + row * rowStride + kbase + hi * 16;
    BF16Frag u;
    u.q[0] = *(const uint4*)p;
    u.q[1] = *(const uint4*)(p + 8);
    return u.v;
}

// B-matrix fragment stored pre-laid-out in LDS: 32 lanes x 16 bf16 contiguous.
__device__ __forceinline__ v16bf ldsB_frag(const __bf16* base, int lane) {
    const __bf16* p = base + lane * 16;
    BF16Frag u;
    u.q[0] = *(const uint4*)p;
    u.q[1] = *(const uint4*)(p + 8);
    return u.v;
}

__device__ __forceinline__ v8f wmma_bf16(v16bf a, v16bf b, v8f c) {
    return __builtin_amdgcn_wmma_f32_16x16x32_bf16(false, a, false, b, (short)0, c, false, false);
}

__global__ __launch_bounds__(256, 1) void cfc_persistent_kernel(
    const float* __restrict__ x,      // [B,T,IN]
    const float* __restrict__ W_bb,   // [260,128]
    const float* __restrict__ b_bb,   // [128]
    const float* __restrict__ W_ff1,  // [128,256]
    const float* __restrict__ b_ff1,  // [256]
    const float* __restrict__ W_ff2,  // [128,256]
    const float* __restrict__ b_ff2,  // [256]
    const float* __restrict__ W_ta,   // [128,256]
    const float* __restrict__ b_ta,   // [256]
    const float* __restrict__ W_tb,   // [128,256]
    const float* __restrict__ b_tb,   // [256]
    const float* __restrict__ W_fc,   // [256,10]
    const float* __restrict__ b_fc,   // [10]
    float* __restrict__ out)          // [B,10]
{
    // LDS buffers
    __shared__ __align__(16) __bf16 sWext[8 * 9 * 512]; // B-fragments of W_ext (73728 B)
    __shared__ __align__(16) __bf16 sA[16 * KEXT];      // A staging: [x(4) | h(256) | pad(28)]
    __shared__ __align__(16) __bf16 sBB[16 * BBD];      // bb activations
    __shared__ __align__(16) __bf16 sH[16 * UNITS];     // h_mean exchange (final)
    __shared__ __align__(16) float  sOut[16 * 16];      // output reduction (final)

    const int tid  = threadIdx.x;
    const int w    = tid >> 5;        // wave id 0..7
    const int lane = tid & 31;
    const int n    = lane & 15;
    const int hi   = lane >> 4;
    const int b0   = blockIdx.x * 16; // batch tile base row

    // ---- init: zero A staging (h0 = 0, pad = 0) ----
    for (int i = tid; i < 16 * KEXT; i += 256) sA[i] = (__bf16)0.0f;

    // ---- init: W_ext B-fragments into LDS (rows 0..3 = x-part, 4..259 = h-part of W_bb) ----
    for (int idx = tid; idx < 8 * 9 * 512; idx += 256) {
        const int frag = idx >> 9;          // nt*9 + kt
        const int rem  = idx & 511;         // l*16 + e
        const int l  = rem >> 4, e = rem & 15;
        const int nt = frag / 9, kt = frag - nt * 9;
        const int ln = l & 15, lh = l >> 4;
        const int j  = nt * 16 + ln;                 // bb unit
        const int k  = kt * 32 + lh * 16 + e;        // K index in [0,288)
        const float v = (k < IN_SZ + UNITS) ? W_bb[k * BBD + j] : 0.0f;
        sWext[idx] = (__bf16)v;
    }

    // ---- init: head weight B-fragments resident in VGPRs ----
    // wave w owns unit columns u in [32w, 32w+32) for heads {ff1, ff2, t=ta+tb}
    v16bf wB2[3][2][4];
#pragma unroll
    for (int hd = 0; hd < 3; ++hd)
#pragma unroll
        for (int nt = 0; nt < 2; ++nt)
#pragma unroll
            for (int kt = 0; kt < 4; ++kt) {
                BF16Frag u;
#pragma unroll
                for (int e = 0; e < 16; ++e) {
                    const int k = kt * 32 + hi * 16 + e;       // 0..127
                    const int uu = w * 32 + nt * 16 + n;       // unit col
                    float v;
                    if (hd == 0)      v = W_ff1[k * UNITS + uu];
                    else if (hd == 1) v = W_ff2[k * UNITS + uu];
                    else              v = W_ta[k * UNITS + uu] + W_tb[k * UNITS + uu];
                    u.e[e] = (__bf16)v;
                }
                wB2[hd][nt][kt] = u.v;
            }

    // ---- biases ----
    const float biasBB = b_bb[w * 16 + n];
    float bF1[2], bF2[2], bT[2];
#pragma unroll
    for (int nt = 0; nt < 2; ++nt) {
        const int uu = w * 32 + nt * 16 + n;
        bF1[nt] = b_ff1[uu];
        bF2[nt] = b_ff2[uu];
        bT[nt]  = b_ta[uu] + b_tb[uu];
    }

    // ---- persistent h accumulator (f32) ----
    v8f hsum[2];
#pragma unroll
    for (int nt = 0; nt < 2; ++nt) hsum[nt] = splat8(0.0f);

    // ---- prefetch x_0 into registers ----
    const int xm = tid >> 2, xi = tid & 3;
    float xreg = 0.0f;
    if (tid < 64) xreg = x[(size_t)(b0 + xm) * T_SZ * IN_SZ + xi];

    __syncthreads();

    // ================= time loop =================
#pragma unroll 1
    for (int t = 0; t < T_SZ; ++t) {
        // stage x_t (prefetched last step) into A cols 0..3
        if (tid < 64) {
            sA[xm * KEXT + xi] = (__bf16)xreg;
        }
        __syncthreads();

        // prefetch x_{t+1}: a full step of compute hides this load
        if (tid < 64 && (t + 1) < T_SZ) {
            xreg = x[((size_t)(b0 + xm) * T_SZ + (t + 1)) * IN_SZ + xi];
        }

        // ---- phase A: bb = lecun_tanh([x|h] @ W_ext + b) ; wave w -> bb cols 16w..16w+15
        // Issue all fragment loads up front, then run the accumulation chain.
        v16bf aF[9], bWx[9];
#pragma unroll
        for (int kt = 0; kt < 9; ++kt) {
            aF[kt]  = ldsA_frag(sA, KEXT, kt * 32, lane);
            bWx[kt] = ldsB_frag(&sWext[(w * 9 + kt) * 512], lane);
        }
        v8f cb = splat8(biasBB);
#pragma unroll
        for (int kt = 0; kt < 9; ++kt) {
            cb = wmma_bf16(aF[kt], bWx[kt], cb);
        }
#pragma unroll
        for (int r = 0; r < 8; ++r) {
            const float vv = 1.7159f * fast_tanh(0.666f * cb[r]);
            sBB[(r + hi * 8) * BBD + (w * 16 + n)] = (__bf16)vv;
        }
        __syncthreads();

        // ---- phase B: three head GEMMs on the same A (bb), gate, update h ----
        v16bf a2[4];
#pragma unroll
        for (int kt = 0; kt < 4; ++kt) a2[kt] = ldsA_frag(sBB, BBD, kt * 32, lane);

#pragma unroll
        for (int nt = 0; nt < 2; ++nt) {
            v8f c0 = splat8(bF1[nt]);
            v8f c1 = splat8(bF2[nt]);
            v8f c2 = splat8(bT[nt]);
#pragma unroll
            for (int kt = 0; kt < 4; ++kt) {
                c0 = wmma_bf16(a2[kt], wB2[0][nt][kt], c0);
                c1 = wmma_bf16(a2[kt], wB2[1][nt][kt], c1);
                c2 = wmma_bf16(a2[kt], wB2[2][nt][kt], c2);
            }
#pragma unroll
            for (int r = 0; r < 8; ++r) {
                const float e1 = fast_tanh(c0[r]);
                const float e2 = fast_tanh(c1[r]);
                const float ti = 0.5f * fast_tanh(0.5f * c2[r]) + 0.5f; // sigmoid
                const float hn = e1 + ti * (e2 - e1);
                hsum[nt][r] += hn;
                // write h into A staging for next step (col 4 + unit)
                sA[(r + hi * 8) * KEXT + 4 + w * 32 + nt * 16 + n] = (__bf16)hn;
            }
        }
        // top-of-loop barrier covers sA/sBB hazards
    }

    // ================= output: (sum_t h / T) @ W_fc + b_fc =================
    __syncthreads();
    const float invT = 1.0f / (float)T_SZ;
#pragma unroll
    for (int nt = 0; nt < 2; ++nt)
#pragma unroll
        for (int r = 0; r < 8; ++r)
            sH[(r + hi * 8) * UNITS + w * 32 + nt * 16 + n] = (__bf16)(hsum[nt][r] * invT);
    sOut[tid] = 0.0f;
    __syncthreads();

    // W_fc B fragment for this wave's K-slice [32w, 32w+32), N padded 10->16
    BF16Frag ub;
#pragma unroll
    for (int e = 0; e < 16; ++e) {
        const int k = w * 32 + hi * 16 + e;
        ub.e[e] = (__bf16)((n < NC) ? W_fc[k * NC + n] : 0.0f);
    }
    v16bf aH = ldsA_frag(sH, UNITS, w * 32, lane);
    v8f co = wmma_bf16(aH, ub.v, splat8(0.0f));
#pragma unroll
    for (int r = 0; r < 8; ++r)
        atomicAdd(&sOut[(r + hi * 8) * 16 + n], co[r]);
    __syncthreads();

    if (tid < 16 * NC) {
        const int m = tid / NC, c = tid - m * NC;
        out[(size_t)(b0 + m) * NC + c] = sOut[m * 16 + c] + b_fc[c];
    }
}

extern "C" void kernel_launch(void* const* d_in, const int* in_sizes, int n_in,
                              void* d_out, int out_size, void* d_ws, size_t ws_size,
                              hipStream_t stream) {
    (void)in_sizes; (void)n_in; (void)d_ws; (void)ws_size; (void)out_size;
    const float* x     = (const float*)d_in[0];
    const float* W_bb  = (const float*)d_in[1];
    const float* b_bb  = (const float*)d_in[2];
    const float* W_ff1 = (const float*)d_in[3];
    const float* b_ff1 = (const float*)d_in[4];
    const float* W_ff2 = (const float*)d_in[5];
    const float* b_ff2 = (const float*)d_in[6];
    const float* W_ta  = (const float*)d_in[7];
    const float* b_ta  = (const float*)d_in[8];
    const float* W_tb  = (const float*)d_in[9];
    const float* b_tb  = (const float*)d_in[10];
    const float* W_fc  = (const float*)d_in[11];
    const float* b_fc  = (const float*)d_in[12];
    float* out = (float*)d_out;

    dim3 grid(B_SZ / 16);   // 64 workgroups, one per 16-row batch tile
    dim3 block(256);        // 8 wave32
    cfc_persistent_kernel<<<grid, block, 0, stream>>>(
        x, W_bb, b_bb, W_ff1, b_ff1, W_ff2, b_ff2,
        W_ta, b_ta, W_tb, b_tb, W_fc, b_fc, out);
}
